// VectorQuantizerHybridV4_72310069395606
// MI455X (gfx1250) — compile-verified, matrix operand order checked by reading, not run on hardware
//
#include <hip/hip_runtime.h>
#include <hip/hip_bf16.h>
#include <math.h>

// Problem sizes (fixed by the reference)
#define BATCH    32768
#define EMB_DIM  512
#define NUM_EMB  8192
#define QSIZE    (BATCH * EMB_DIM)
#define NTILES   (NUM_EMB / 16)

typedef __attribute__((ext_vector_type(16))) __bf16          v16bf;
typedef __attribute__((ext_vector_type(8)))  float           v8f;
typedef __attribute__((ext_vector_type(8)))  unsigned short  v8u16;

union BF16Frag {
  v16bf          bf;
  unsigned short u[16];
  v8u16          h[2];
};

__device__ __forceinline__ unsigned short f32_to_bf16_rne(float f) {
  unsigned int u = __float_as_uint(f);
  unsigned int lsb = (u >> 16) & 1u;
  u += 0x7FFFu + lsb;              // round-to-nearest-even
  return (unsigned short)(u >> 16);
}

// ---------------------------------------------------------------------------
// Kernel 1: W fp32 -> bf16 (row major) + per-row squared norm in fp32.
// grid = NUM_EMB blocks, 128 threads (4 elems/thread).
// ---------------------------------------------------------------------------
__global__ void vq_prep_w(const float* __restrict__ W,
                          unsigned short* __restrict__ Wb,
                          float* __restrict__ wnorm) {
  const int row = blockIdx.x;
  const int tid = threadIdx.x;
  const float* wr = W + (size_t)row * EMB_DIM;
  float4 v = ((const float4*)wr)[tid];

  ushort4 o;
  o.x = f32_to_bf16_rne(v.x);
  o.y = f32_to_bf16_rne(v.y);
  o.z = f32_to_bf16_rne(v.z);
  o.w = f32_to_bf16_rne(v.w);
  ((ushort4*)(Wb + (size_t)row * EMB_DIM))[tid] = o;

  float s = v.x * v.x + v.y * v.y + v.z * v.z + v.w * v.w;
  #pragma unroll
  for (int m = 16; m >= 1; m >>= 1) s += __shfl_down(s, m, 32);
  __shared__ float ls[4];
  if ((tid & 31) == 0) ls[tid >> 5] = s;
  __syncthreads();
  if (tid == 0) wnorm[row] = ls[0] + ls[1] + ls[2] + ls[3];
}

// ---------------------------------------------------------------------------
// Kernel 2: zero the assignment histogram.
// ---------------------------------------------------------------------------
__global__ void vq_zero_counts(int* __restrict__ counts) {
  const int i = blockIdx.x * blockDim.x + threadIdx.x;
  if (i < NUM_EMB) counts[i] = 0;
}

// ---------------------------------------------------------------------------
// Kernel 3: fused bf16-WMMA distance GEMM + row argmin.
// 256 threads = 8 waves; each wave owns 16 z rows with the full 16x512 bf16
// A tile register-resident (16 fragments x 8 VGPRs; launch_bounds(256,1)
// grants the full VGPR budget so nothing spills). Each iteration handles TWO
// 16-code N-tiles with independent accumulator chains so B loads of one chain
// overlap the WMMAs of the other. Epilogue: d = |W_n|^2 - 2*dot with a
// running per-lane argmin in the native C/D register layout.
// grid = BATCH / 128 = 256 blocks.
// ---------------------------------------------------------------------------
__global__ void __launch_bounds__(256, 1)
vq_argmin(const float* __restrict__ z,
          const unsigned short* __restrict__ Wb,
          const float* __restrict__ wnorm,
          int* __restrict__ idx_out) {
  const int lane  = threadIdx.x & 31;
  const int wave  = threadIdx.x >> 5;
  const int mbase = (blockIdx.x * 8 + wave) * 16;
  const int mrow  = mbase + (lane & 15);
  const int kb    = (lane >> 4) * 8;           // K phase: 0 or 8

  // --- Preload A fragments (z tile), converting fp32 -> bf16 in-register ---
  BF16Frag a[16];
  const float* zr = z + (size_t)mrow * EMB_DIM;
  #pragma unroll
  for (int f = 0; f < 16; ++f) {
    const int k0 = f * 32 + kb;
    float4 x0 = *(const float4*)(zr + k0);
    float4 x1 = *(const float4*)(zr + k0 + 4);
    float4 y0 = *(const float4*)(zr + k0 + 16);
    float4 y1 = *(const float4*)(zr + k0 + 20);
    a[f].u[0]  = f32_to_bf16_rne(x0.x);  a[f].u[1]  = f32_to_bf16_rne(x0.y);
    a[f].u[2]  = f32_to_bf16_rne(x0.z);  a[f].u[3]  = f32_to_bf16_rne(x0.w);
    a[f].u[4]  = f32_to_bf16_rne(x1.x);  a[f].u[5]  = f32_to_bf16_rne(x1.y);
    a[f].u[6]  = f32_to_bf16_rne(x1.z);  a[f].u[7]  = f32_to_bf16_rne(x1.w);
    a[f].u[8]  = f32_to_bf16_rne(y0.x);  a[f].u[9]  = f32_to_bf16_rne(y0.y);
    a[f].u[10] = f32_to_bf16_rne(y0.z);  a[f].u[11] = f32_to_bf16_rne(y0.w);
    a[f].u[12] = f32_to_bf16_rne(y1.x);  a[f].u[13] = f32_to_bf16_rne(y1.y);
    a[f].u[14] = f32_to_bf16_rne(y1.z);  a[f].u[15] = f32_to_bf16_rne(y1.w);
  }

  float minv[8];
  int   mini[8];
  #pragma unroll
  for (int r = 0; r < 8; ++r) { minv[r] = 3.4e38f; mini[r] = 0; }

  // Stagger the N sweep per block to spread L2 traffic over the 8MB W_bf16.
  const int NP = NTILES / 2;                   // 256 tile-pairs
  const int tstart = blockIdx.x & (NP - 1);

  for (int t = 0; t < NP; ++t) {
    const int tp    = ((t + tstart) & (NP - 1)) * 2;
    const int nbase = tp * 16;
    const int n0    = nbase + (lane & 15);
    const int n1    = n0 + 16;
    const unsigned short* wr0 = Wb + (size_t)n0 * EMB_DIM + kb;
    const unsigned short* wr1 = wr0 + 16 * EMB_DIM;

    v8f acc0 = {};
    v8f acc1 = {};
    #pragma unroll
    for (int f = 0; f < 16; ++f) {
      BF16Frag b0, b1;
      b0.h[0] = *(const v8u16*)(wr0 + f * 32);
      b0.h[1] = *(const v8u16*)(wr0 + f * 32 + 16);
      b1.h[0] = *(const v8u16*)(wr1 + f * 32);
      b1.h[1] = *(const v8u16*)(wr1 + f * 32 + 16);
      acc0 = __builtin_amdgcn_wmma_f32_16x16x32_bf16(
          false, a[f].bf, false, b0.bf, (short)0, acc0, false, false);
      acc1 = __builtin_amdgcn_wmma_f32_16x16x32_bf16(
          false, a[f].bf, false, b1.bf, (short)0, acc1, false, false);
    }

    const float wn0 = wnorm[n0];
    const float wn1 = wnorm[n1];
    #pragma unroll
    for (int r = 0; r < 8; ++r) {
      const float d0 = wn0 - 2.0f * acc0[r];
      if (d0 < minv[r]) { minv[r] = d0; mini[r] = n0; }
      const float d1 = wn1 - 2.0f * acc1[r];
      if (d1 < minv[r]) { minv[r] = d1; mini[r] = n1; }
    }
  }

  // Cross-lane argmin within each 16-lane half (matches C/D lane split);
  // tie-break on the lower index like jnp.argmin.
  #pragma unroll
  for (int r = 0; r < 8; ++r) {
    float v = minv[r];
    int   i = mini[r];
    #pragma unroll
    for (int m = 8; m >= 1; m >>= 1) {
      const float ov = __shfl_xor(v, m, 32);
      const int   oi = __shfl_xor(i, m, 32);
      if (ov < v || (ov == v && oi < i)) { v = ov; i = oi; }
    }
    minv[r] = v; mini[r] = i;
  }

  if ((lane & 15) == 0) {
    const int mofs = (lane >> 4) * 8;   // lanes 0-15 -> rows 0..7, 16-31 -> 8..15
    #pragma unroll
    for (int r = 0; r < 8; ++r)
      idx_out[mbase + mofs + r] = mini[r];
  }
}

// ---------------------------------------------------------------------------
// Kernel 4: gather quantized rows (fp32 W), per-row SSE partial (deterministic,
// no float atomics), histogram increment, indices as float to d_out.
// grid = BATCH blocks, 128 threads.
// ---------------------------------------------------------------------------
__global__ void vq_gather_loss(const float* __restrict__ z,
                               const float* __restrict__ W,
                               const int* __restrict__ idx,
                               float* __restrict__ out_q,
                               float* __restrict__ out_idx_f,
                               float* __restrict__ sse_part,
                               int* __restrict__ counts) {
  const int row = blockIdx.x;
  const int tid = threadIdx.x;
  const int k   = idx[row];

  float4 q  = ((const float4*)(W + (size_t)k   * EMB_DIM))[tid];
  float4 zz = ((const float4*)(z + (size_t)row * EMB_DIM))[tid];
  ((float4*)(out_q + (size_t)row * EMB_DIM))[tid] = q;   // quantized_st == quantized

  const float d0 = zz.x - q.x, d1 = zz.y - q.y, d2 = zz.z - q.z, d3 = zz.w - q.w;
  float s = d0 * d0 + d1 * d1 + d2 * d2 + d3 * d3;
  #pragma unroll
  for (int m = 16; m >= 1; m >>= 1) s += __shfl_down(s, m, 32);
  __shared__ float ls[4];
  if ((tid & 31) == 0) ls[tid >> 5] = s;
  __syncthreads();
  if (tid == 0) {
    sse_part[row]  = ls[0] + ls[1] + ls[2] + ls[3];
    out_idx_f[row] = (float)k;
    atomicAdd(&counts[k], 1);                 // integer: order-independent
  }
}

// ---------------------------------------------------------------------------
// Kernel 5: deterministic final reduction: sum SSE partials, entropy of the
// assignment histogram, combined scalar loss.
// ---------------------------------------------------------------------------
__global__ void vq_finalize(const float* __restrict__ sse_part,
                            const int* __restrict__ counts,
                            float* __restrict__ out_loss) {
  const int tid = threadIdx.x;   // 256
  float s = 0.0f;
  for (int i = tid; i < BATCH; i += 256) s += sse_part[i];
  float e = 0.0f;
  for (int i = tid; i < NUM_EMB; i += 256) {
    const float p = (float)counts[i] / (float)BATCH;
    e -= p * logf(p + 1e-10f);
  }
  #pragma unroll
  for (int m = 16; m >= 1; m >>= 1) {
    s += __shfl_down(s, m, 32);
    e += __shfl_down(e, m, 32);
  }
  __shared__ float rs[8], re[8];
  if ((tid & 31) == 0) { rs[tid >> 5] = s; re[tid >> 5] = e; }
  __syncthreads();
  if (tid == 0) {
    float S = 0.0f, E = 0.0f;
    #pragma unroll
    for (int w = 0; w < 8; ++w) { S += rs[w]; E += re[w]; }
    const float mse    = S / (float)QSIZE;
    const float maxent = logf((float)NUM_EMB);
    // codebook_loss + 0.25*commitment_loss = 1.25*mse (identical values)
    out_loss[0] = 1.25f * mse + 0.1f * (1.0f - E / maxent);
  }
}

// ---------------------------------------------------------------------------
// Launch. d_in[0]=z (32768x512 f32), d_in[1]=W (8192x512 f32).
// d_out layout: [quantized_st 32768*512][loss 1][indices-as-float 32768].
// ---------------------------------------------------------------------------
extern "C" void kernel_launch(void* const* d_in, const int* in_sizes, int n_in,
                              void* d_out, int out_size, void* d_ws, size_t ws_size,
                              hipStream_t stream) {
  const float* z = (const float*)d_in[0];
  const float* W = (const float*)d_in[1];
  float* out = (float*)d_out;

  char* ws = (char*)d_ws;
  const size_t WB_BYTES = (size_t)NUM_EMB * EMB_DIM * 2;  // 8 MiB bf16 codebook
  unsigned short* Wb    = (unsigned short*)ws;
  float*          wnorm = (float*)(ws + WB_BYTES);
  int*            cnts  = (int*)  (ws + WB_BYTES + 32768);
  float*          ssep  = (float*)(ws + WB_BYTES + 65536);
  int*            idxw  = (int*)  (ws + WB_BYTES + 65536 + (size_t)BATCH * 4);

  float* out_q    = out;
  float* out_loss = out + (size_t)QSIZE;
  float* out_idx  = out + (size_t)QSIZE + 1;

  vq_prep_w<<<NUM_EMB, 128, 0, stream>>>(W, Wb, wnorm);
  vq_zero_counts<<<NUM_EMB / 256, 256, 0, stream>>>(cnts);
  vq_argmin<<<BATCH / 128, 256, 0, stream>>>(z, Wb, wnorm, idxw);
  vq_gather_loss<<<BATCH, 128, 0, stream>>>(z, W, idxw, out_q, out_idx, ssep, cnts);
  vq_finalize<<<1, 256, 0, stream>>>(ssep, cnts, out_loss);
}
